// MultiHeadCrossAttention_12463995093183
// MI455X (gfx1250) — matrix operand order repaired and checked by reading, not compiled, not run on hardware
//
#include <hip/hip_runtime.h>
#include <hip/hip_bf16.h>

// ---------------------------------------------------------------------------
// Fused MHA block for MI455X (gfx1250, wave32, WMMA 16x16x32 f16->f32)
// Double-buffered LDS tiles, fragment-major B layout (all fragment traffic is
// ds_load_b128), batched fragment preloads so WMMAs issue back-to-back.
// ---------------------------------------------------------------------------

#define BATCH   4
#define S_LEN   2048
#define DM      768
#define HEADS   12
#define DK      64
#define MROWS   (BATCH * S_LEN)   // 8192
#define LN_EPS  1e-5f

typedef __attribute__((ext_vector_type(16))) _Float16 v16h;
typedef __attribute__((ext_vector_type(8)))  _Float16 v8h;
typedef __attribute__((ext_vector_type(4)))  _Float16 v4h;
typedef __attribute__((ext_vector_type(8)))  float    v8f;

__device__ __forceinline__ v8f wmma_f16(v16h a, v16h b, v8f c) {
    return __builtin_amdgcn_wmma_f32_16x16x32_f16(
        false, a, false, b, (short)0, c, false, false);
}

__device__ __forceinline__ v16h combine16(v8h lo, v8h hi) {
    v16h r;
#pragma unroll
    for (int e = 0; e < 8; ++e) { r[e] = lo[e]; r[e + 8] = hi[e]; }
    return r;
}

// A fragment (16x32 f16): two contiguous 16B chunks per lane.
__device__ __forceinline__ v16h load_frag_a(const _Float16* base, int lda,
                                            int m0, int k0) {
    const int lane = threadIdx.x & 31;
    const int m    = m0 + (lane & 15);
    const int ks   = (lane >> 4) * 16;
    const _Float16* p = base + (size_t)m * lda + k0 + ks;
    v8h lo = *(const v8h*)p;
    v8h hi = *(const v8h*)(p + 8);
    return combine16(lo, hi);
}

// B fragment from fragment-major LDS: chunkPair -> [2][32][8] halves.
__device__ __forceinline__ v16h load_frag_b_fm(const _Float16* cp) {
    const int lane = threadIdx.x & 31;
    v8h lo = *(const v8h*)(cp + lane * 8);
    v8h hi = *(const v8h*)(cp + 256 + lane * 8);
    return combine16(lo, hi);
}

// ---------------------------------------------------------------------------
// Kernel 0: f32 -> f16 conversion (weights)
// ---------------------------------------------------------------------------
__global__ void cvt_f32_f16_kernel(const float* __restrict__ src,
                                   _Float16* __restrict__ dst, int n4) {
    int i = blockIdx.x * blockDim.x + threadIdx.x;
    if (i < n4) {
        float4 f = ((const float4*)src)[i];
        v4h h;
        h[0] = (_Float16)f.x; h[1] = (_Float16)f.y;
        h[2] = (_Float16)f.z; h[3] = (_Float16)f.w;
        *(v4h*)&dst[(size_t)i * 4] = h;
    }
}

// ---------------------------------------------------------------------------
// Kernel 1: projection GEMM  out = X(f32) @ W(f16) + bias, split heads.
//   mode 0: out[b,h,s,d]   mode 1: out[b,h,d,s] (K transposed)
// Block 256 thr (8 waves), tile 256x64 (32 rows/wave), BK=32, double-buffered.
// Grid (32, 12).
// ---------------------------------------------------------------------------
__global__ void gemm_proj_kernel(const float* __restrict__ X,
                                 const _Float16* __restrict__ W16,
                                 const float* __restrict__ bias,
                                 _Float16* __restrict__ out, int mode) {
    __shared__ __align__(16) _Float16 As[2][256][40];
    __shared__ __align__(16) _Float16 Bf[2][4][2][32][8];

    const int tid    = threadIdx.x;
    const int wave   = tid >> 5;
    const int lane   = tid & 31;
    const int half   = lane >> 4;
    const int m_base = blockIdx.x * 256;
    const int n_base = blockIdx.y * 64;  // one head

    auto fill = [&](int buf, int kb) {
        {   // A: 256x32 f32 -> f16, one row (32 floats) per thread
            const float4* s4 =
                (const float4*)(X + (size_t)(m_base + tid) * DM + kb);
#pragma unroll
            for (int c = 0; c < 2; ++c) {
                float4 f0 = s4[c * 4 + 0], f1 = s4[c * 4 + 1];
                float4 f2 = s4[c * 4 + 2], f3 = s4[c * 4 + 3];
                v8h lo, hi;
                lo[0]=(_Float16)f0.x; lo[1]=(_Float16)f0.y; lo[2]=(_Float16)f0.z; lo[3]=(_Float16)f0.w;
                lo[4]=(_Float16)f1.x; lo[5]=(_Float16)f1.y; lo[6]=(_Float16)f1.z; lo[7]=(_Float16)f1.w;
                hi[0]=(_Float16)f2.x; hi[1]=(_Float16)f2.y; hi[2]=(_Float16)f2.z; hi[3]=(_Float16)f2.w;
                hi[4]=(_Float16)f3.x; hi[5]=(_Float16)f3.y; hi[6]=(_Float16)f3.z; hi[7]=(_Float16)f3.w;
                *(v8h*)&As[buf][tid][c * 16]     = lo;
                *(v8h*)&As[buf][tid][c * 16 + 8] = hi;
            }
        }
        {   // B: 32x64 f16 -> fragment-major scatter
            const int k = tid >> 3, n0 = (tid & 7) * 8;
            v8h w = *(const v8h*)&W16[(size_t)(kb + k) * DM + n_base + n0];
            const int e = k & 15, c = e >> 3, j = e & 7, hk = (k >> 4) & 1;
#pragma unroll
            for (int q = 0; q < 8; ++q) {
                const int n = n0 + q;
                Bf[buf][n >> 4][c][hk * 16 + (n & 15)][j] = w[q];
            }
        }
    };

    v8f acc[2][4] = {};

    fill(0, 0);
    __syncthreads();
    const int NSTEP = DM / 32;  // 24
    for (int i = 0; i < NSTEP; ++i) {
        const int buf = i & 1;
        if (i + 1 < NSTEP) fill(buf ^ 1, (i + 1) * 32);

        v16h af0 = load_frag_a(&As[buf][0][0], 40, wave * 32, 0);
        v16h af1 = load_frag_a(&As[buf][0][0], 40, wave * 32 + 16, 0);
        v16h bfr[4];
#pragma unroll
        for (int t = 0; t < 4; ++t)
            bfr[t] = load_frag_b_fm(&Bf[buf][t][0][0][0]);
#pragma unroll
        for (int t = 0; t < 4; ++t) {
            acc[0][t] = wmma_f16(af0, bfr[t], acc[0][t]);
            acc[1][t] = wmma_f16(af1, bfr[t], acc[1][t]);
        }
        __syncthreads();
    }

    const int h = blockIdx.y;
#pragma unroll
    for (int sub = 0; sub < 2; ++sub)
#pragma unroll
        for (int t = 0; t < 4; ++t)
#pragma unroll
            for (int r = 0; r < 8; ++r) {
                int gm = m_base + wave * 32 + sub * 16 + r + half * 8;
                int d  = t * 16 + (lane & 15);
                float val = acc[sub][t][r] + bias[n_base + d];
                int b = gm >> 11, s = gm & (S_LEN - 1);
                if (mode == 1)
                    out[(((size_t)(b * HEADS + h) * DK + d) * S_LEN) + s] = (_Float16)val;
                else
                    out[(((size_t)(b * HEADS + h) * S_LEN + s) * DK) + d] = (_Float16)val;
            }
}

// ---------------------------------------------------------------------------
// Kernel 2: flash attention. Grid (S/128, B*H), block 256 (8 waves).
// Double-buffered fragment-major K/V tiles; per-wave P strip needs no
// barrier (same-wave LDS ops are in-order). One barrier per key tile.
// ---------------------------------------------------------------------------
__global__ void flash_attn_kernel(const _Float16* __restrict__ Qh,
                                  const _Float16* __restrict__ Kt,
                                  const _Float16* __restrict__ Vh,
                                  _Float16* __restrict__ ctx) {
    __shared__ __align__(16) _Float16 Kf[2][16][32][8];
    __shared__ __align__(16) _Float16 Vf[2][16][32][8];
    __shared__ __align__(16) _Float16 Ps[8][16][40];

    const int tid  = threadIdx.x;
    const int wave = tid >> 5;
    const int lane = tid & 31;
    const int half = lane >> 4;
    const int bh   = blockIdx.y;
    const int q0   = blockIdx.x * 128 + wave * 16;

    const _Float16* Qb  = Qh + (size_t)bh * S_LEN * DK;
    const _Float16* Ktb = Kt + (size_t)bh * DK * S_LEN;
    const _Float16* Vb  = Vh + (size_t)bh * S_LEN * DK;

    const v16h aq0 = load_frag_a(Qb + (size_t)q0 * DK, DK, 0, 0);
    const v16h aq1 = load_frag_a(Qb + (size_t)q0 * DK, DK, 0, 32);

    auto fill = [&](int buf, int kb) {
        {   // K^T tile: rows d (k-dim), cols kpos (n)
            const int d = tid >> 2, c0 = (tid & 3) * 16;
            const _Float16* kp = Ktb + (size_t)d * S_LEN + kb + c0;
            v8h ka = *(const v8h*)kp;
            v8h kz = *(const v8h*)(kp + 8);
            const int e = d & 15, c = e >> 3, j = e & 7;
            const int hd = (d >> 4) & 1, kc = d >> 5;
#pragma unroll
            for (int q = 0; q < 8; ++q) {
                int n = c0 + q;
                Kf[buf][(n >> 4) * 4 + kc * 2 + c][hd * 16 + (n & 15)][j] = ka[q];
            }
#pragma unroll
            for (int q = 0; q < 8; ++q) {
                int n = c0 + 8 + q;
                Kf[buf][(n >> 4) * 4 + kc * 2 + c][hd * 16 + (n & 15)][j] = kz[q];
            }
        }
        {   // V tile: rows kpos (k-dim), cols d (n)
            const int kr = tid >> 2, d0 = (tid & 3) * 16;
            const _Float16* vp = Vb + (size_t)(kb + kr) * DK + d0;
            v8h va = *(const v8h*)vp;
            v8h vz = *(const v8h*)(vp + 8);
            const int e = kr & 15, c = e >> 3, j = e & 7;
            const int hk = (kr >> 4) & 1, kc = kr >> 5;
#pragma unroll
            for (int q = 0; q < 8; ++q) {
                int d = d0 + q;
                Vf[buf][(d >> 4) * 4 + kc * 2 + c][hk * 16 + (d & 15)][j] = va[q];
            }
#pragma unroll
            for (int q = 0; q < 8; ++q) {
                int d = d0 + 8 + q;
                Vf[buf][(d >> 4) * 4 + kc * 2 + c][hk * 16 + (d & 15)][j] = vz[q];
            }
        }
    };

    float mrun[8], lrun[8];
#pragma unroll
    for (int r = 0; r < 8; ++r) { mrun[r] = -1e30f; lrun[r] = 0.f; }
    v8f oacc[4] = {};

    fill(0, 0);
    __syncthreads();
    const int NKB = S_LEN / 64;  // 32
    for (int i = 0; i < NKB; ++i) {
        const int buf = i & 1;
        if (i + 1 < NKB) fill(buf ^ 1, (i + 1) * 64);

        // ---- scores S = Q K^T / sqrt(dk), 16 x 64 f32 ----
        v8f sc[4] = {};
#pragma unroll
        for (int tp = 0; tp < 2; ++tp) {
            const int t0 = tp * 2, t1 = tp * 2 + 1;
            v16h b00 = load_frag_b_fm(&Kf[buf][t0 * 4 + 0][0][0]);
            v16h b01 = load_frag_b_fm(&Kf[buf][t0 * 4 + 2][0][0]);
            v16h b10 = load_frag_b_fm(&Kf[buf][t1 * 4 + 0][0][0]);
            v16h b11 = load_frag_b_fm(&Kf[buf][t1 * 4 + 2][0][0]);
            sc[t0] = wmma_f16(aq0, b00, sc[t0]);
            sc[t0] = wmma_f16(aq1, b01, sc[t0]);
            sc[t1] = wmma_f16(aq0, b10, sc[t1]);
            sc[t1] = wmma_f16(aq1, b11, sc[t1]);
        }
        const float scale = 0.125f;
#pragma unroll
        for (int t = 0; t < 4; ++t)
#pragma unroll
            for (int r = 0; r < 8; ++r) sc[t][r] *= scale;

        // ---- online softmax ----
        float rmax[8];
#pragma unroll
        for (int r = 0; r < 8; ++r)
            rmax[r] = fmaxf(fmaxf(sc[0][r], sc[1][r]), fmaxf(sc[2][r], sc[3][r]));
#pragma unroll
        for (int off = 1; off <= 8; off <<= 1)
#pragma unroll
            for (int r = 0; r < 8; ++r)
                rmax[r] = fmaxf(rmax[r], __shfl_xor(rmax[r], off, 32));

        float mnew[8], alpha[8], psum[8];
#pragma unroll
        for (int r = 0; r < 8; ++r) {
            mnew[r]  = fmaxf(mrun[r], rmax[r]);
            alpha[r] = __expf(mrun[r] - mnew[r]);
            psum[r]  = 0.f;
        }
#pragma unroll
        for (int t = 0; t < 4; ++t)
#pragma unroll
            for (int r = 0; r < 8; ++r) {
                float p = __expf(sc[t][r] - mnew[r]);
                sc[t][r] = p;
                psum[r] += p;
                oacc[t][r] *= alpha[r];
            }
#pragma unroll
        for (int off = 1; off <= 8; off <<= 1)
#pragma unroll
            for (int r = 0; r < 8; ++r) psum[r] += __shfl_xor(psum[r], off, 32);
#pragma unroll
        for (int r = 0; r < 8; ++r) {
            lrun[r] = lrun[r] * alpha[r] + psum[r];
            mrun[r] = mnew[r];
        }

        // ---- P transpose through per-wave LDS strip (same-wave order) ----
#pragma unroll
        for (int t = 0; t < 4; ++t)
#pragma unroll
            for (int r = 0; r < 8; ++r)
                Ps[wave][r + half * 8][t * 16 + (lane & 15)] = (_Float16)sc[t][r];

        const v16h ap0 = load_frag_a(&Ps[wave][0][0], 40, 0, 0);
        const v16h ap1 = load_frag_a(&Ps[wave][0][0], 40, 0, 32);
#pragma unroll
        for (int tp = 0; tp < 2; ++tp) {
            const int t0 = tp * 2, t1 = tp * 2 + 1;
            v16h b00 = load_frag_b_fm(&Vf[buf][t0 * 4 + 0][0][0]);
            v16h b01 = load_frag_b_fm(&Vf[buf][t0 * 4 + 2][0][0]);
            v16h b10 = load_frag_b_fm(&Vf[buf][t1 * 4 + 0][0][0]);
            v16h b11 = load_frag_b_fm(&Vf[buf][t1 * 4 + 2][0][0]);
            oacc[t0] = wmma_f16(ap0, b00, oacc[t0]);
            oacc[t0] = wmma_f16(ap1, b01, oacc[t0]);
            oacc[t1] = wmma_f16(ap0, b10, oacc[t1]);
            oacc[t1] = wmma_f16(ap1, b11, oacc[t1]);
        }
        __syncthreads();
    }

    const int b = bh / HEADS, h = bh % HEADS;
#pragma unroll
    for (int t = 0; t < 4; ++t)
#pragma unroll
        for (int r = 0; r < 8; ++r) {
            int q = q0 + r + half * 8;
            int d = t * 16 + (lane & 15);
            float val = oacc[t][r] / lrun[r];
            ctx[((size_t)(b * S_LEN + q)) * DM + h * DK + d] = (_Float16)val;
        }
}

// ---------------------------------------------------------------------------
// Kernel 3: output projection + bias + residual:  x = resid + ctx @ W_O + b_O
// Same 256x64 double-buffered structure as kernel 1 (f16 A input, f32 out).
// ---------------------------------------------------------------------------
__global__ void gemm_out_kernel(const _Float16* __restrict__ ctx,
                                const _Float16* __restrict__ Wo16,
                                const float* __restrict__ bias,
                                const float* __restrict__ resid,
                                float* __restrict__ Xout) {
    __shared__ __align__(16) _Float16 As[2][256][40];
    __shared__ __align__(16) _Float16 Bf[2][4][2][32][8];

    const int tid    = threadIdx.x;
    const int wave   = tid >> 5;
    const int lane   = tid & 31;
    const int half   = lane >> 4;
    const int m_base = blockIdx.x * 256;
    const int n_base = blockIdx.y * 64;

    auto fill = [&](int buf, int kb) {
        {   // A: 256x32 f16 copy, one row per thread
            const _Float16* src = ctx + (size_t)(m_base + tid) * DM + kb;
#pragma unroll
            for (int c = 0; c < 4; ++c)
                *(v8h*)&As[buf][tid][c * 8] = *(const v8h*)(src + c * 8);
        }
        {   // B scatter to fragment-major
            const int k = tid >> 3, n0 = (tid & 7) * 8;
            v8h w = *(const v8h*)&Wo16[(size_t)(kb + k) * DM + n_base + n0];
            const int e = k & 15, c = e >> 3, j = e & 7, hk = (k >> 4) & 1;
#pragma unroll
            for (int q = 0; q < 8; ++q) {
                const int n = n0 + q;
                Bf[buf][n >> 4][c][hk * 16 + (n & 15)][j] = w[q];
            }
        }
    };

    v8f acc[2][4] = {};

    fill(0, 0);
    __syncthreads();
    const int NSTEP = DM / 32;
    for (int i = 0; i < NSTEP; ++i) {
        const int buf = i & 1;
        if (i + 1 < NSTEP) fill(buf ^ 1, (i + 1) * 32);

        v16h af0 = load_frag_a(&As[buf][0][0], 40, wave * 32, 0);
        v16h af1 = load_frag_a(&As[buf][0][0], 40, wave * 32 + 16, 0);
        v16h bfr[4];
#pragma unroll
        for (int t = 0; t < 4; ++t)
            bfr[t] = load_frag_b_fm(&Bf[buf][t][0][0][0]);
#pragma unroll
        for (int t = 0; t < 4; ++t) {
            acc[0][t] = wmma_f16(af0, bfr[t], acc[0][t]);
            acc[1][t] = wmma_f16(af1, bfr[t], acc[1][t]);
        }
        __syncthreads();
    }

#pragma unroll
    for (int sub = 0; sub < 2; ++sub)
#pragma unroll
        for (int t = 0; t < 4; ++t)
#pragma unroll
            for (int r = 0; r < 8; ++r) {
                int gm = m_base + wave * 32 + sub * 16 + r + half * 8;
                int gn = n_base + t * 16 + (lane & 15);
                Xout[(size_t)gm * DM + gn] =
                    acc[sub][t][r] + bias[gn] + resid[(size_t)gm * DM + gn];
            }
}

// ---------------------------------------------------------------------------
// Kernel 4: row LayerNorm (768 cols, 256 thr -> 3 elems/thread)
// ---------------------------------------------------------------------------
__global__ void layernorm_kernel(const float* __restrict__ X,
                                 const float* __restrict__ gamma,
                                 const float* __restrict__ beta,
                                 float* __restrict__ out) {
    __shared__ float red[16];
    const int row  = blockIdx.x;
    const int tid  = threadIdx.x;
    const int wave = tid >> 5;
    const int lane = tid & 31;
    const float* xr = X + (size_t)row * DM;

    float v0 = xr[tid], v1 = xr[tid + 256], v2 = xr[tid + 512];
    float s1 = v0 + v1 + v2;
    float s2 = v0 * v0 + v1 * v1 + v2 * v2;
#pragma unroll
    for (int off = 1; off <= 16; off <<= 1) {
        s1 += __shfl_xor(s1, off, 32);
        s2 += __shfl_xor(s2, off, 32);
    }
    if (lane == 0) { red[wave] = s1; red[8 + wave] = s2; }
    __syncthreads();
    float t1 = 0.f, t2 = 0.f;
#pragma unroll
    for (int w = 0; w < 8; ++w) { t1 += red[w]; t2 += red[8 + w]; }
    const float mean = t1 * (1.0f / DM);
    const float var  = t2 * (1.0f / DM) - mean * mean;
    const float rstd = rsqrtf(var + LN_EPS);

    float* orow = out + (size_t)row * DM;
    orow[tid]       = (v0 - mean) * rstd * gamma[tid]       + beta[tid];
    orow[tid + 256] = (v1 - mean) * rstd * gamma[tid + 256] + beta[tid + 256];
    orow[tid + 512] = (v2 - mean) * rstd * gamma[tid + 512] + beta[tid + 512];
}

// ---------------------------------------------------------------------------
extern "C" void kernel_launch(void* const* d_in, const int* in_sizes, int n_in,
                              void* d_out, int out_size, void* d_ws, size_t ws_size,
                              hipStream_t stream) {
    const float* Q_in  = (const float*)d_in[0];
    const float* K_in  = (const float*)d_in[1];
    const float* V_in  = (const float*)d_in[2];
    const float* W_Q   = (const float*)d_in[3];
    const float* b_Q   = (const float*)d_in[4];
    const float* W_K   = (const float*)d_in[5];
    const float* b_K   = (const float*)d_in[6];
    const float* W_V   = (const float*)d_in[7];
    const float* b_V   = (const float*)d_in[8];
    const float* W_O   = (const float*)d_in[9];
    const float* b_O   = (const float*)d_in[10];
    const float* gamma = (const float*)d_in[11];
    const float* beta  = (const float*)d_in[12];
    float* out = (float*)d_out;

    char* base = (char*)d_ws;
    size_t off = 0;
    auto carve = [&](size_t bytes) {
        char* p = base + off;
        off = (off + bytes + 255) & ~(size_t)255;
        return p;
    };
    const size_t W16_ELEMS = (size_t)DM * DM;
    const size_t HQ_ELEMS  = (size_t)BATCH * HEADS * S_LEN * DK;
    const size_t CTX_ELEMS = (size_t)MROWS * DM;

    _Float16* wq16 = (_Float16*)carve(W16_ELEMS * 2);
    _Float16* wk16 = (_Float16*)carve(W16_ELEMS * 2);
    _Float16* wv16 = (_Float16*)carve(W16_ELEMS * 2);
    _Float16* wo16 = (_Float16*)carve(W16_ELEMS * 2);
    _Float16* Qh   = (_Float16*)carve(HQ_ELEMS * 2);
    _Float16* Kt   = (_Float16*)carve(HQ_ELEMS * 2);
    _Float16* Vh   = (_Float16*)carve(HQ_ELEMS * 2);
    _Float16* ctx  = (_Float16*)carve(CTX_ELEMS * 2);
    float*    Xres = (float*)   carve(CTX_ELEMS * 4);
    (void)ws_size; (void)in_sizes; (void)n_in; (void)out_size;

    const int n4 = (int)(W16_ELEMS / 4);
    const int cvt_blocks = (n4 + 255) / 256;
    cvt_f32_f16_kernel<<<cvt_blocks, 256, 0, stream>>>(W_Q, wq16, n4);
    cvt_f32_f16_kernel<<<cvt_blocks, 256, 0, stream>>>(W_K, wk16, n4);
    cvt_f32_f16_kernel<<<cvt_blocks, 256, 0, stream>>>(W_V, wv16, n4);
    cvt_f32_f16_kernel<<<cvt_blocks, 256, 0, stream>>>(W_O, wo16, n4);

    dim3 gproj(MROWS / 256, DM / 64);
    gemm_proj_kernel<<<gproj, 256, 0, stream>>>(Q_in, wq16, b_Q, Qh, 0);
    gemm_proj_kernel<<<gproj, 256, 0, stream>>>(K_in, wk16, b_K, Kt, 1);
    gemm_proj_kernel<<<gproj, 256, 0, stream>>>(V_in, wv16, b_V, Vh, 0);

    dim3 gattn(S_LEN / 128, BATCH * HEADS);
    flash_attn_kernel<<<gattn, 256, 0, stream>>>(Qh, Kt, Vh, ctx);

    gemm_out_kernel<<<gproj, 256, 0, stream>>>(ctx, wo16, b_O, Q_in, Xres);

    layernorm_kernel<<<MROWS, 256, 0, stream>>>(Xres, gamma, beta, out);
}
